// CNNGRUNet_21852793602352
// MI455X (gfx1250) — matrix operand order repaired and checked by reading, not compiled, not run on hardware
//
#include <hip/hip_runtime.h>
#include <hip/hip_bf16.h>
#include <math.h>

// ---------------------------------------------------------------------------
// Problem constants (from reference): L=512, T=96, C=H=150, K=5
//   Padded: channels 150 -> 160, reduction K*160 = 800 = 25 * 32 (WMMA K=32)
//   Output cols 3*160 = 480 = 30 tiles of 16; rows 512 = 32 tiles of 16.
// ---------------------------------------------------------------------------
#define LB   512
#define TT   96
#define CH   150
#define CP   160          // padded channels
#define KW   5
#define KT   25           // (5*160)/32 k-tiles
#define NC   480          // padded 3H
#define NT   30           // 480/16 col tiles
#define BT   32           // 512/16 row tiles
#define HROWS (LB + 4)    // 516: h / input rows with +-2 halo
#define WTILE (32 * 16)   // halves per packed weight (kt,nt) tile

typedef __attribute__((ext_vector_type(16))) _Float16 v16h;
typedef __attribute__((ext_vector_type(8)))  _Float16 v8h;
typedef __attribute__((ext_vector_type(8)))  float    v8f;

union AFrag { v16h v; v8h h[2]; };
union CFrag { v8f  v; float f[8]; };

// ---------------------------------------------------------------------------
// Pack xs (512,96,150) f32 -> inF16 [516][96][160] f16 with zero halo/pad.
// Row rb in [0,516) corresponds to b = rb-2.
// ---------------------------------------------------------------------------
__global__ void __launch_bounds__(256) pack_x_kernel(
    const float* __restrict__ xs, _Float16* __restrict__ inF16)
{
    int idx = blockIdx.x * 256 + threadIdx.x;
    const int total = HROWS * TT * CP;
    if (idx >= total) return;
    int j  = idx % CP;
    int t  = (idx / CP) % TT;
    int rb = idx / (CP * TT);
    int b  = rb - 2;
    float v = 0.0f;
    if (b >= 0 && b < LB && j < CH)
        v = xs[((size_t)b * TT + t) * CH + j];
    inF16[idx] = (_Float16)v;
}

__global__ void __launch_bounds__(256) zero_f16_kernel(_Float16* __restrict__ p, int n)
{
    int idx = blockIdx.x * 256 + threadIdx.x;
    if (idx < n) p[idx] = (_Float16)0.0f;
}

// ---------------------------------------------------------------------------
// Pack conv weights W (5,150,450) f32 into B-fragment order:
//   wp[((nt*KT + kt)*32 + lane)*16 + i]  = W[k][j][g*150+o]  (f16, 0 padded)
// where r = kt*32 + (lane>>4)*16 + i = k*160 + j  (reduction index)
//       n = nt*16 + (lane&15)        = g*160 + o  (output column)
// Each WMMA B operand is then 16 contiguous halves per lane.
// ---------------------------------------------------------------------------
__global__ void __launch_bounds__(256) pack_w_kernel(
    const float* __restrict__ W, _Float16* __restrict__ wp)
{
    int idx = blockIdx.x * 256 + threadIdx.x;
    const int total = NT * KT * 32 * 16;
    if (idx >= total) return;
    int i   = idx & 15;
    int l   = (idx >> 4) & 31;
    int kt  = (idx >> 9) % KT;
    int nt  = (idx >> 9) / KT;
    int r   = kt * 32 + ((l >> 4) & 1) * 16 + i;
    int n   = nt * 16 + (l & 15);
    int k = r / CP, j = r % CP;
    int g = n / CP, o = n % CP;
    float v = 0.0f;
    if (j < CH && o < CH)
        v = W[((size_t)k * CH + j) * (3 * CH) + g * CH + o];
    wp[idx] = (_Float16)v;
}

// ---------------------------------------------------------------------------
// Zero the recurrent state (both ping-pong f16 halo buffers + f32 state).
// ---------------------------------------------------------------------------
__global__ void __launch_bounds__(256) init_state_kernel(
    _Float16* __restrict__ hb0, _Float16* __restrict__ hb1, float* __restrict__ h32)
{
    int idx = blockIdx.x * 256 + threadIdx.x;
    if (idx < HROWS * CP) { hb0[idx] = (_Float16)0.0f; hb1[idx] = (_Float16)0.0f; }
    if (idx < LB * CP)    h32[idx] = 0.0f;
}

// ---------------------------------------------------------------------------
// Phase A: xi[t][b][0:480] = conv_b(input, Wi) + bias    (parallel over t)
// One wave per (t, row-tile-pair, col-tile-pair): 2x2 register tile.
// Per k-step: 2 A frags + 2 B frags (8 x b128 loads) feed 4 WMMAs
// -> 2 loads/WMMA (vs 4 for an untiled wave).
// A fragment (16-bit, 16x32): lane l = row m=l&15; halves 0..15 -> K=Kb+0..15,
// Kb=(l>>4)*16.  r = k*160+j -> inF16[(b+k)][t][j];
// 32 | 160 so runs never straddle k: two contiguous 16B loads.
// ---------------------------------------------------------------------------
__global__ void __launch_bounds__(32) xi_conv_kernel(
    const _Float16* __restrict__ inF16,  // [516][96][160]
    const _Float16* __restrict__ wpack,  // [NT][KT][32][16]
    const float*    __restrict__ bias,   // [450]
    _Float16*       __restrict__ xi)     // [96][512][480]
{
    int bid = blockIdx.x;
    int t   = bid / ((BT / 2) * (NT / 2));
    int r2  = bid % ((BT / 2) * (NT / 2));
    int bt2 = r2 / (NT / 2);             // row-tile pair: rows bt2*32 .. +31
    int nt2 = r2 % (NT / 2);             // col-tile pair: tiles 2*nt2, 2*nt2+1
    int l   = threadIdx.x;
    int m   = l & 15;
    int kb  = (l >> 4) * 16;
    int b0  = bt2 * 32 + m;
    int b1  = b0 + 16;
    int nt0 = nt2 * 2;

    v8f acc00 = {}, acc01 = {}, acc10 = {}, acc11 = {};
    AFrag A0, A1, B0, B1;
    const _Float16* wp0 = wpack + ((size_t)(nt0 * KT) * 32 + l) * 16;
    const _Float16* wp1 = wp0 + (size_t)KT * WTILE;
    #pragma unroll
    for (int kt = 0; kt < KT; ++kt) {
        const size_t io = (size_t)(kt / 5) * (TT * CP) + (kt % 5) * 32 + kb;
        const _Float16* p0 = inF16 + ((size_t)b0 * TT + t) * CP + io;
        const _Float16* p1 = inF16 + ((size_t)b1 * TT + t) * CP + io;
        A0.h[0] = *(const v8h*)(p0);   A0.h[1] = *(const v8h*)(p0 + 8);
        A1.h[0] = *(const v8h*)(p1);   A1.h[1] = *(const v8h*)(p1 + 8);
        B0.h[0] = *(const v8h*)(wp0);  B0.h[1] = *(const v8h*)(wp0 + 8);
        B1.h[0] = *(const v8h*)(wp1);  B1.h[1] = *(const v8h*)(wp1 + 8);
        wp0 += WTILE; wp1 += WTILE;
        acc00 = __builtin_amdgcn_wmma_f32_16x16x32_f16(false, A0.v, false, B0.v, (short)0, acc00, false, false);
        acc01 = __builtin_amdgcn_wmma_f32_16x16x32_f16(false, A0.v, false, B1.v, (short)0, acc01, false, false);
        acc10 = __builtin_amdgcn_wmma_f32_16x16x32_f16(false, A1.v, false, B0.v, (short)0, acc10, false, false);
        acc11 = __builtin_amdgcn_wmma_f32_16x16x32_f16(false, A1.v, false, B1.v, (short)0, acc11, false, false);
    }

    int N  = l & 15;
    int Mb = (l >> 4) * 8;
    v8f accs[4] = { acc00, acc01, acc10, acc11 };
    #pragma unroll
    for (int q = 0; q < 4; ++q) {
        int nt  = nt0 + (q & 1);
        int col = nt * 16 + N;
        int g = col / CP, o = col % CP;
        float bv = (o < CH) ? bias[g * CH + o] : 0.0f;
        int rbase = bt2 * 32 + (q >> 1) * 16 + Mb;
        CFrag C; C.v = accs[q];
        #pragma unroll
        for (int v = 0; v < 8; ++v)
            xi[((size_t)t * LB + rbase + v) * NC + col] = (_Float16)(C.f[v] + bv);
    }
}

// ---------------------------------------------------------------------------
// Phase B: one timestep.  32 workgroups (one per 16-row tile, keeps all WGPs
// busy in the serial phase) x 15 waves; each wave owns TWO col tiles
// (1x2 register tile: 6 loads / 2 WMMAs, A fragment reused).
// hh -> LDS, then all 480 threads fuse gates + state update.
// h ping-pong buffers carry the +-2 halo; the kernel-launch boundary is the
// device-wide sync between timesteps.
// ---------------------------------------------------------------------------
__global__ void __launch_bounds__(480) gru_step_kernel(
    const _Float16* __restrict__ hprev,  // [516][160]
    _Float16*       __restrict__ hnext,  // [516][160]
    float*          __restrict__ h32,    // [512][160]
    const _Float16* __restrict__ whp,    // packed Wh
    const _Float16* __restrict__ xi,     // [96][512][480]
    _Float16*       __restrict__ out16,  // layer0 -> next layer's inF16 [516][96][160] (or null)
    float*          __restrict__ out32,  // layer1 -> d_out [512][96][150] (or null)
    int t)
{
    __shared__ float hh[16 * NC];        // 30720 B

    int bt   = blockIdx.x;
    int wave = threadIdx.x >> 5;         // 0..14
    int l    = threadIdx.x & 31;
    int nt0  = wave * 2;
    int m    = l & 15;
    int kb   = (l >> 4) * 16;
    int b    = bt * 16 + m;

    v8f acc0 = {}, acc1 = {};
    AFrag A, B0, B1;
    const _Float16* wp0 = whp + ((size_t)(nt0 * KT) * 32 + l) * 16;
    const _Float16* wp1 = wp0 + (size_t)KT * WTILE;
    #pragma unroll
    for (int kt = 0; kt < KT; ++kt) {
        const _Float16* p = hprev + (size_t)(b + kt / 5) * CP + (kt % 5) * 32 + kb;
        A.h[0]  = *(const v8h*)(p);    A.h[1]  = *(const v8h*)(p + 8);
        B0.h[0] = *(const v8h*)(wp0);  B0.h[1] = *(const v8h*)(wp0 + 8);
        B1.h[0] = *(const v8h*)(wp1);  B1.h[1] = *(const v8h*)(wp1 + 8);
        wp0 += WTILE; wp1 += WTILE;
        acc0 = __builtin_amdgcn_wmma_f32_16x16x32_f16(false, A.v, false, B0.v, (short)0, acc0, false, false);
        acc1 = __builtin_amdgcn_wmma_f32_16x16x32_f16(false, A.v, false, B1.v, (short)0, acc1, false, false);
    }
    {
        int N = l & 15, Mb = (l >> 4) * 8;
        CFrag C0; C0.v = acc0;
        CFrag C1; C1.v = acc1;
        #pragma unroll
        for (int v = 0; v < 8; ++v) {
            hh[(Mb + v) * NC + nt0 * 16 + N]        = C0.f[v];
            hh[(Mb + v) * NC + (nt0 + 1) * 16 + N]  = C1.f[v];
        }
    }
    __syncthreads();

    const _Float16* xib = xi + ((size_t)t * LB + bt * 16) * NC;
    for (int e = threadIdx.x; e < 16 * CP; e += 480) {
        int mm = e / CP, j = e % CP;
        int bb = bt * 16 + mm;
        float hnew = 0.0f;
        if (j < CH) {
            float xr = (float)xib[mm * NC + j];
            float xz = (float)xib[mm * NC + CP + j];
            float xn = (float)xib[mm * NC + 2 * CP + j];
            float hr = hh[mm * NC + j];
            float hz = hh[mm * NC + CP + j];
            float hn = hh[mm * NC + 2 * CP + j];
            float r  = 1.0f / (1.0f + __expf(-(xr + hr)));
            float z  = 1.0f / (1.0f + __expf(-(xz + hz)));
            float a  = xn + r * hn;
            float ex = __expf(2.0f * a);
            float n  = (ex - 1.0f) / (ex + 1.0f);   // tanh
            float ho = h32[bb * CP + j];
            hnew = (1.0f - z) * n + z * ho;
        }
        h32[bb * CP + j] = hnew;
        hnext[(size_t)(bb + 2) * CP + j] = (_Float16)hnew;
        if (out16) out16[((size_t)(bb + 2) * TT + t) * CP + j] = (_Float16)hnew;
        if (out32 && j < CH) out32[((size_t)bb * TT + t) * CH + j] = hnew;
    }
}

// ---------------------------------------------------------------------------
// Host: workspace carve-up + launch schedule
// ---------------------------------------------------------------------------
extern "C" void kernel_launch(void* const* d_in, const int* in_sizes, int n_in,
                              void* d_out, int out_size, void* d_ws, size_t ws_size,
                              hipStream_t stream)
{
    const float* xs  = (const float*)d_in[0];
    const float* Wi0 = (const float*)d_in[1];
    const float* bi0 = (const float*)d_in[2];
    const float* Wh0 = (const float*)d_in[3];
    const float* Wi1 = (const float*)d_in[4];
    const float* bi1 = (const float*)d_in[5];
    const float* Wh1 = (const float*)d_in[6];
    float* out = (float*)d_out;

    char* ws = (char*)d_ws;
    size_t off = 0;
    auto carve = [&](size_t bytes) {
        void* p = ws + off;
        off = (off + bytes + 255) & ~(size_t)255;
        return p;
    };
    const size_t inF16Bytes = (size_t)HROWS * TT * CP * 2;          // 15.85 MB
    const size_t xiBytes    = (size_t)TT * LB * NC * 2;             // 47.2 MB
    const size_t wpBytes    = (size_t)NT * KT * 32 * 16 * 2;        // 0.77 MB
    const size_t hbBytes    = (size_t)HROWS * CP * 2;
    const size_t h32Bytes   = (size_t)LB * CP * 4;

    _Float16* in0  = (_Float16*)carve(inF16Bytes);
    _Float16* in1  = (_Float16*)carve(inF16Bytes);
    _Float16* xi   = (_Float16*)carve(xiBytes);
    _Float16* wip0 = (_Float16*)carve(wpBytes);
    _Float16* whp0 = (_Float16*)carve(wpBytes);
    _Float16* wip1 = (_Float16*)carve(wpBytes);
    _Float16* whp1 = (_Float16*)carve(wpBytes);
    _Float16* hb0  = (_Float16*)carve(hbBytes);
    _Float16* hb1  = (_Float16*)carve(hbBytes);
    float*    h32  = (float*)carve(h32Bytes);

    const int packXN = HROWS * TT * CP;
    const int packWN = NT * KT * 32 * 16;
    const int initN  = HROWS * CP;
    const int phaseABlocks = TT * (BT / 2) * (NT / 2);   // 23040 waves

    // ---- prep ----
    pack_x_kernel<<<(packXN + 255) / 256, 256, 0, stream>>>(xs, in0);
    zero_f16_kernel<<<(packXN + 255) / 256, 256, 0, stream>>>(in1, packXN);
    pack_w_kernel<<<(packWN + 255) / 256, 256, 0, stream>>>(Wi0, wip0);
    pack_w_kernel<<<(packWN + 255) / 256, 256, 0, stream>>>(Wh0, whp0);
    pack_w_kernel<<<(packWN + 255) / 256, 256, 0, stream>>>(Wi1, wip1);
    pack_w_kernel<<<(packWN + 255) / 256, 256, 0, stream>>>(Wh1, whp1);

    // ---- layer 0 ----
    xi_conv_kernel<<<phaseABlocks, 32, 0, stream>>>(in0, wip0, bi0, xi);
    init_state_kernel<<<(initN + 255) / 256, 256, 0, stream>>>(hb0, hb1, h32);
    for (int t = 0; t < TT; ++t) {
        _Float16* hp = (t & 1) ? hb1 : hb0;
        _Float16* hn = (t & 1) ? hb0 : hb1;
        gru_step_kernel<<<BT, 480, 0, stream>>>(hp, hn, h32, whp0, xi,
                                                in1, (float*)nullptr, t);
    }

    // ---- layer 1 ----
    xi_conv_kernel<<<phaseABlocks, 32, 0, stream>>>(in1, wip1, bi1, xi);
    init_state_kernel<<<(initN + 255) / 256, 256, 0, stream>>>(hb0, hb1, h32);
    for (int t = 0; t < TT; ++t) {
        _Float16* hp = (t & 1) ? hb1 : hb0;
        _Float16* hn = (t & 1) ? hb0 : hb1;
        gru_step_kernel<<<BT, 480, 0, stream>>>(hp, hn, h32, whp1, xi,
                                                (_Float16*)nullptr, out, t);
    }
}
